// attention_62242666054436
// MI455X (gfx1250) — compile-verified, matrix operand order checked by reading, not run on hardware
//
#include <hip/hip_runtime.h>

// ---------------------------------------------------------------------------
// CDNA5 (gfx1250) fused attention layer: QKV GEMM + RoPE -> flash attention
// -> output GEMM, all on v_wmma_f32_16x16x32_bf16 (f32 accumulate).
// All matrix-fragment LDS reads are contiguous (ds_load_b128 pairs).
// bf16 tiles staged with GLOBAL_LOAD_ASYNC_TO_LDS_B128 when available.
// ---------------------------------------------------------------------------

typedef __attribute__((ext_vector_type(16))) __bf16 v16bf;
typedef __attribute__((ext_vector_type(8)))  float  v8f;

union BFrag { v16bf v; unsigned short u[16]; };

#if defined(__has_builtin)
#if __has_builtin(__builtin_amdgcn_global_load_async_to_lds_b128) && \
    __has_builtin(__builtin_amdgcn_s_wait_asynccnt)
#define USE_ASYNC_LDS 1
#endif
#endif

typedef int v4i __attribute__((vector_size(16)));
typedef __attribute__((address_space(1))) v4i g_v4i;   // global int4
typedef __attribute__((address_space(3))) v4i l_v4i;   // LDS int4

__device__ __forceinline__ void copy_b128_to_lds(void* lds_dst, const void* gsrc) {
#ifdef USE_ASYNC_LDS
  __builtin_amdgcn_global_load_async_to_lds_b128((g_v4i*)gsrc, (l_v4i*)lds_dst,
                                                 0, 0);
#else
  *(uint4*)lds_dst = *(const uint4*)gsrc;
#endif
}
__device__ __forceinline__ void async_wait() {
#ifdef USE_ASYNC_LDS
  __builtin_amdgcn_s_wait_asynccnt(0);
#endif
}

// fp32 -> bf16, round-to-nearest-even
__device__ __forceinline__ unsigned short f2bf(float f) {
  unsigned u = __builtin_bit_cast(unsigned, f);
  unsigned r = (u + 0x7FFFu + ((u >> 16) & 1u)) >> 16;
  return (unsigned short)r;
}

// K-index inside a 16-bit 16x32 fragment, per ISA 7.12.2:
// lanes 0-15: elems 0-7 -> K0-7,  elems 8-15 -> K16-23
// lanes 16-31: elems 0-7 -> K8-15, elems 8-15 -> K24-31
__device__ __forceinline__ int kidx(int e, int half) {
  return ((e & 8) << 1) + (half << 3) + (e & 7);
}

// Fragment whose 16 elements are base[row*ld + kidx(e)]; with K contiguous in
// memory this lowers to 2x ds_load_b128.
__device__ __forceinline__ v16bf load_frag(const unsigned short* base, int row,
                                           int ld, int half) {
  BFrag f;
#pragma unroll
  for (int e = 0; e < 16; ++e) f.u[e] = base[row * ld + kidx(e, half)];
  return f.v;
}

__device__ __forceinline__ v8f wmma_bf16(v16bf a, v16bf b, v8f c) {
  return __builtin_amdgcn_wmma_f32_16x16x32_bf16(false, a, false, b, (short)0, c,
                                                 false, false);
}

__device__ __forceinline__ float redmax16(float v) {
#pragma unroll
  for (int m = 1; m < 16; m <<= 1) v = fmaxf(v, __shfl_xor(v, m, 32));
  return v;
}
__device__ __forceinline__ float redsum16(float v) {
#pragma unroll
  for (int m = 1; m < 16; m <<= 1) v += __shfl_xor(v, m, 32);
  return v;
}

#define HID   2048
#define MROWS 8192   // B*S
#define SEQ   2048
#define HDIM  128

// ---------------------------------------------------------------------------
// Kernel 1: X @ [Wq|Wk|Wv] with fused RoPE. Block tile 128(M) x 64(N), 8 waves
// (2x4), K-step 32. fp32 sources converted to bf16 while staging into LDS.
// A tile row-major [128][32]; B tile transposed [64 cols][32 k] so both
// operand fragments are contiguous.
// q,k stored [B,H,S,D]; v stored TRANSPOSED [B,KV,D,S] (free in the epilogue:
// a C-fragment lane owns one column and 8 consecutive rows -> uint4 store).
// ---------------------------------------------------------------------------
__global__ __launch_bounds__(256) void qkv_gemm_rope(
    const float* __restrict__ x, const float* __restrict__ cosb,
    const float* __restrict__ sinb, const float* __restrict__ Wq,
    const float* __restrict__ Wk, const float* __restrict__ Wv,
    unsigned short* __restrict__ qb, unsigned short* __restrict__ kb,
    unsigned short* __restrict__ vb) {
  __shared__ __align__(16) unsigned short As[128 * 32];
  __shared__ __align__(16) unsigned short Bt[64 * 32];

  const int tid = threadIdx.x;
  const int lane = tid & 31, wave = tid >> 5;
  const int half = lane >> 4, l16 = lane & 15;
  const int wm = wave >> 2, wn = wave & 3;
  const int rowBase = blockIdx.y * 128;
  const int nBase = blockIdx.x * 64;

  const float* W;
  int ldw, colLoc;
  if (nBase < 2048)      { W = Wq; ldw = 2048; colLoc = nBase; }
  else if (nBase < 2560) { W = Wk; ldw = 512;  colLoc = nBase - 2048; }
  else                   { W = Wv; ldw = 512;  colLoc = nBase - 2560; }

  v8f acc[4] = {};

  for (int k0 = 0; k0 < HID; k0 += 32) {
    __syncthreads();
    // A tile: 128x32 fp32 -> bf16 (row-major)
#pragma unroll
    for (int it = 0; it < 4; ++it) {
      int idx = tid + it * 256;           // 0..1023 float4 chunks
      int r = idx >> 3, c4 = (idx & 7) * 4;
      float4 f = *(const float4*)(x + (size_t)(rowBase + r) * HID + k0 + c4);
      unsigned p0 = (unsigned)f2bf(f.x) | ((unsigned)f2bf(f.y) << 16);
      unsigned p1 = (unsigned)f2bf(f.z) | ((unsigned)f2bf(f.w) << 16);
      *(unsigned*)&As[r * 32 + c4]     = p0;
      *(unsigned*)&As[r * 32 + c4 + 2] = p1;
    }
    // B tile: 32x64 fp32 -> bf16, written transposed [col][k]
#pragma unroll
    for (int it = 0; it < 2; ++it) {
      int idx = tid + it * 256;           // 0..511 float4 chunks
      int r = idx >> 4, c4 = (idx & 15) * 4;
      float4 f = *(const float4*)(W + (size_t)(k0 + r) * ldw + colLoc + c4);
      Bt[(c4 + 0) * 32 + r] = f2bf(f.x);
      Bt[(c4 + 1) * 32 + r] = f2bf(f.y);
      Bt[(c4 + 2) * 32 + r] = f2bf(f.z);
      Bt[(c4 + 3) * 32 + r] = f2bf(f.w);
    }
    __syncthreads();

    v16bf bf = load_frag(Bt, wn * 16 + l16, 32, half);
#pragma unroll
    for (int mf = 0; mf < 4; ++mf) {
      v16bf af = load_frag(As + (wm * 64 + mf * 16) * 32, l16, 32, half);
      acc[mf] = wmma_bf16(af, bf, acc[mf]);
    }
  }

  // Epilogue: RoPE (even/odd pair lives in xor-1 lane) + scatter
  const int col = nBase + wn * 16 + l16;
#pragma unroll
  for (int mf = 0; mf < 4; ++mf) {
    const int s0 = rowBase + wm * 64 + mf * 16 + half * 8;  // 8 consecutive s
    const int b = s0 >> 11, sb = s0 & (SEQ - 1);
    float part[8];
#pragma unroll
    for (int g = 0; g < 8; ++g) part[g] = __shfl_xor(acc[mf][g], 1, 32);

    if (col < 2560) {                         // Q or K: RoPE + scattered b16
      const int isQ = (col < 2048) ? 1 : 0;
      const int c2 = isQ ? col : (col - 2048);
      const int hh = c2 >> 7, d = c2 & 127;
#pragma unroll
      for (int g = 0; g < 8; ++g) {
        int s = sb + g;
        float c = cosb[s * HDIM + d], sn = sinb[s * HDIM + d];
        float val = acc[mf][g];
        float r2 = (d & 1) ? (val * c + part[g] * sn) : (val * c - part[g] * sn);
        if (isQ)
          qb[(((size_t)(b * 16 + hh)) * SEQ + s) * HDIM + d] = f2bf(r2);
        else
          kb[(((size_t)(b * 4 + hh)) * SEQ + s) * HDIM + d] = f2bf(r2);
      }
    } else {                                  // V: packed transposed store
      const int c2 = col - 2560;
      const int kv = c2 >> 7, d = c2 & 127;
      unsigned short h8[8];
#pragma unroll
      for (int g = 0; g < 8; ++g) h8[g] = f2bf(acc[mf][g]);
      uint4 pk;
      pk.x = (unsigned)h8[0] | ((unsigned)h8[1] << 16);
      pk.y = (unsigned)h8[2] | ((unsigned)h8[3] << 16);
      pk.z = (unsigned)h8[4] | ((unsigned)h8[5] << 16);
      pk.w = (unsigned)h8[6] | ((unsigned)h8[7] << 16);
      *(uint4*)&vb[(((size_t)(b * 4 + kv)) * HDIM + d) * SEQ + sb] = pk;
    }
  }
}

// ---------------------------------------------------------------------------
// Kernel 2: causal flash attention. One wave per 16-row Q tile per (b,h).
// 32-key tiles: QK^T (8 wmma) -> online softmax -> P.V (8 wmma).
// K staged [key][d] (contiguous d), V staged [d][key] from the transposed
// global layout (contiguous keys) -> every fragment is 2x ds_load_b128.
// ---------------------------------------------------------------------------
__global__ __launch_bounds__(32) void flash_attn(
    const unsigned short* __restrict__ qb, const unsigned short* __restrict__ kb,
    const unsigned short* __restrict__ vb, unsigned short* __restrict__ ao) {
  __shared__ __align__(16) unsigned short Qs[16 * 128];
  __shared__ __align__(16) unsigned short Ks[32 * 128];
  __shared__ __align__(16) unsigned short Vt[128 * 32];
  __shared__ __align__(16) unsigned short Ps[16 * 32];

  const int lane = threadIdx.x;
  const int half = lane >> 4, l16 = lane & 15;
  const int qbase = blockIdx.x * 16;
  const int bh = blockIdx.y;
  const int b = bh >> 4, h = bh & 15, kv = h >> 2;
  const float SCALE = 0.08838834764831845f;   // 1/sqrt(128)

  const unsigned short* qsrc = qb + (((size_t)(b * 16 + h)) * SEQ + qbase) * HDIM;
  const unsigned short* ks0  = kb + ((size_t)(b * 4 + kv)) * SEQ * HDIM;
  const unsigned short* vt0  = vb + ((size_t)(b * 4 + kv)) * HDIM * SEQ;

  // stage Q (16x128 bf16 = 256 x b128 chunks)
#pragma unroll
  for (int it = 0; it < 8; ++it) {
    int idx = lane + it * 32;
    copy_b128_to_lds(&Qs[idx * 8], qsrc + (size_t)idx * 8);
  }
  async_wait();
  __syncthreads();

  v16bf qf[4];
#pragma unroll
  for (int dt = 0; dt < 4; ++dt)
    qf[dt] = load_frag(Qs + dt * 32, l16, 128, half);

  v8f o[8] = {};
  float mi[8], li[8];
#pragma unroll
  for (int g = 0; g < 8; ++g) { mi[g] = -INFINITY; li[g] = 0.f; }

  const int nkt = (qbase + 47) >> 5;          // causal tile bound
  for (int kt = 0; kt < nkt; ++kt) {
    __syncthreads();
    // K tile: 32 keys x 128 d, row-major
    const unsigned short* ks = ks0 + (size_t)kt * 32 * HDIM;
#pragma unroll
    for (int it = 0; it < 16; ++it) {
      int idx = lane + it * 32;               // 512 chunks of 8
      copy_b128_to_lds(&Ks[idx * 8], ks + (size_t)idx * 8);
    }
    // V tile: 128 d x 32 keys from transposed global layout
#pragma unroll
    for (int it = 0; it < 16; ++it) {
      int idx = lane + it * 32;               // 512 chunks of 8
      int d = idx >> 2, c8 = (idx & 3) * 8;
      copy_b128_to_lds(&Vt[d * 32 + c8], vt0 + (size_t)d * SEQ + kt * 32 + c8);
    }
    async_wait();
    __syncthreads();

    // scores: Q(16x128) @ K^T(128x32) -> 2 fragments of 16x16
    v8f sc[2] = {};
#pragma unroll
    for (int dt = 0; dt < 4; ++dt) {
#pragma unroll
      for (int nt = 0; nt < 2; ++nt) {
        v16bf bk = load_frag(Ks + dt * 32, nt * 16 + l16, 128, half);
        sc[nt] = wmma_bf16(qf[dt], bk, sc[nt]);
      }
    }

    // online softmax (rows live in elems g + 8*half; cols across 16 lanes)
    float pv[2][8], rm[8];
#pragma unroll
    for (int g = 0; g < 8; ++g) rm[g] = -INFINITY;
#pragma unroll
    for (int nt = 0; nt < 2; ++nt) {
#pragma unroll
      for (int g = 0; g < 8; ++g) {
        int j = kt * 32 + nt * 16 + l16;
        int qi = qbase + g + half * 8;
        float v = sc[nt][g] * SCALE;
        v = (j <= qi) ? v : -1e30f;
        pv[nt][g] = v;
        rm[g] = fmaxf(rm[g], v);
      }
    }
    float al[8];
#pragma unroll
    for (int g = 0; g < 8; ++g) {
      float rmax = redmax16(rm[g]);
      float mn = fmaxf(mi[g], rmax);
      al[g] = __expf(mi[g] - mn);
      mi[g] = mn;
      float s0 = 0.f;
#pragma unroll
      for (int nt = 0; nt < 2; ++nt) {
        float p = __expf(pv[nt][g] - mn);
        pv[nt][g] = p;
        s0 += p;
      }
      s0 = redsum16(s0);
      li[g] = li[g] * al[g] + s0;
    }
#pragma unroll
    for (int f = 0; f < 8; ++f)
#pragma unroll
      for (int g = 0; g < 8; ++g) o[f][g] *= al[g];

    // P (C layout) -> LDS -> A layout for P @ V
#pragma unroll
    for (int nt = 0; nt < 2; ++nt)
#pragma unroll
      for (int g = 0; g < 8; ++g)
        Ps[(g + half * 8) * 32 + nt * 16 + l16] = f2bf(pv[nt][g]);
    __syncthreads();

    v16bf pf = load_frag(Ps, l16, 32, half);
#pragma unroll
    for (int f = 0; f < 8; ++f) {
      v16bf bv = load_frag(Vt, f * 16 + l16, 32, half);
      o[f] = wmma_bf16(pf, bv, o[f]);
    }
  }

  // epilogue: O /= l, store bf16 to attn_out[b*S+s, h*128+d]
#pragma unroll
  for (int f = 0; f < 8; ++f) {
#pragma unroll
    for (int g = 0; g < 8; ++g) {
      int s = qbase + g + half * 8;
      float val = o[f][g] / li[g];
      ao[((size_t)(b * SEQ + s)) * HID + h * HDIM + f * 16 + l16] = f2bf(val);
    }
  }
}

// ---------------------------------------------------------------------------
// Kernel 3: attn_out(bf16) @ Wo(fp32->bf16) -> fp32 out. Same tiling as K1;
// bf16 A tile staged straight to LDS (async path when available).
// ---------------------------------------------------------------------------
__global__ __launch_bounds__(256) void out_gemm(
    const unsigned short* __restrict__ ao, const float* __restrict__ Wo,
    float* __restrict__ out) {
  __shared__ __align__(16) unsigned short As[128 * 32];
  __shared__ __align__(16) unsigned short Bt[64 * 32];

  const int tid = threadIdx.x;
  const int lane = tid & 31, wave = tid >> 5;
  const int half = lane >> 4, l16 = lane & 15;
  const int wm = wave >> 2, wn = wave & 3;
  const int rowBase = blockIdx.y * 128;
  const int nBase = blockIdx.x * 64;

  v8f acc[4] = {};

  for (int k0 = 0; k0 < HID; k0 += 32) {
    __syncthreads();
#pragma unroll
    for (int it = 0; it < 2; ++it) {         // A: 128x32 bf16 = 512 x b128
      int idx = tid + it * 256;
      int r = idx >> 2, c8 = (idx & 3) * 8;
      copy_b128_to_lds(&As[r * 32 + c8],
                       ao + (size_t)(rowBase + r) * HID + k0 + c8);
    }
#pragma unroll
    for (int it = 0; it < 2; ++it) {         // B: 32x64 fp32 -> bf16, transposed
      int idx = tid + it * 256;
      int r = idx >> 4, c4 = (idx & 15) * 4;
      float4 f = *(const float4*)(Wo + (size_t)(k0 + r) * HID + nBase + c4);
      Bt[(c4 + 0) * 32 + r] = f2bf(f.x);
      Bt[(c4 + 1) * 32 + r] = f2bf(f.y);
      Bt[(c4 + 2) * 32 + r] = f2bf(f.z);
      Bt[(c4 + 3) * 32 + r] = f2bf(f.w);
    }
    async_wait();
    __syncthreads();

    v16bf bf = load_frag(Bt, wn * 16 + l16, 32, half);
#pragma unroll
    for (int mf = 0; mf < 4; ++mf) {
      v16bf af = load_frag(As + (wm * 64 + mf * 16) * 32, l16, 32, half);
      acc[mf] = wmma_bf16(af, bf, acc[mf]);
    }
  }

  const int col = nBase + wn * 16 + l16;
#pragma unroll
  for (int mf = 0; mf < 4; ++mf)
#pragma unroll
    for (int g = 0; g < 8; ++g) {
      int row = rowBase + wm * 64 + mf * 16 + g + half * 8;
      out[(size_t)row * HID + col] = acc[mf][g];
    }
}

// ---------------------------------------------------------------------------
extern "C" void kernel_launch(void* const* d_in, const int* in_sizes, int n_in,
                              void* d_out, int out_size, void* d_ws, size_t ws_size,
                              hipStream_t stream) {
  const float* x    = (const float*)d_in[0];
  const float* cosb = (const float*)d_in[1];
  const float* sinb = (const float*)d_in[2];
  const float* Wq   = (const float*)d_in[3];
  const float* Wk   = (const float*)d_in[4];
  const float* Wv   = (const float*)d_in[5];
  const float* Wo   = (const float*)d_in[6];
  float* out = (float*)d_out;

  unsigned short* qb = (unsigned short*)d_ws;            // 16 Mi bf16
  unsigned short* kb = qb + (size_t)MROWS * HID;         //  4 Mi bf16
  unsigned short* vb = kb + (size_t)4 * 4 * SEQ * HDIM;  //  4 Mi bf16 [B,KV,D,S]
  unsigned short* ao = vb + (size_t)4 * 4 * SEQ * HDIM;  // 16 Mi bf16

  qkv_gemm_rope<<<dim3(48, 64), 256, 0, stream>>>(x, cosb, sinb, Wq, Wk, Wv,
                                                  qb, kb, vb);
  flash_attn<<<dim3(128, 64), 32, 0, stream>>>(qb, kb, vb, ao);
  out_gemm<<<dim3(32, 64), 256, 0, stream>>>(ao, Wo, out);
}
